// BatchMAFLayer_84971632984479
// MI455X (gfx1250) — matrix operand order repaired and checked by reading, not compiled, not run on hardware
//
#include <hip/hip_runtime.h>
#include <hip/hip_bf16.h>

namespace {

constexpr int D_    = 32;    // autoregressive dims
constexpr int NNET  = 64;    // independent MADE networks
constexpr int HID   = 256;
constexpr int BTOT  = 4096;  // batch rows
constexpr int F_    = 64;    // x feature stride (we use first 32)
constexpr int TB    = 128;   // rows per block
constexpr int NTHR  = 256;   // 8 waves

// LDS row strides (elements), chosen 16B-aligned and bank-conflict-friendly
constexpr int XFLD = 36;     // x fp32 tile      (TB x 36 f32)    = 18432 B
constexpr int XLD  = 40;     // x bf16 tile      (TB x 40 bf16)   = 10240 B
constexpr int HLD  = 264;    // h bf16 tile      (TB x 264 bf16)  = 67584 B
constexpr int W1LD = 40;     // W1^T [256 n][32 k]
constexpr int W2LD = 72;     // W2^T chunk [256 n][64 k]          = 36864 B (max)
constexpr int W3LD = 264;    // W3^T [64 n][256 k]                = 33792 B
constexpr int OLD  = 68;     // out fp32 tile [TB][64]            = 34816 B
constexpr int WBUF_BYTES = 36864;

constexpr float HALF_LOG_2PI = 0.9189385332046727f;

typedef __attribute__((ext_vector_type(16))) __bf16 v16bf;
typedef __attribute__((ext_vector_type(8)))  float  v8f;

// A fragment (16x32 bf16, M=rows, K=cols) from row-major LDS, stride ld.
// Lane m=lane&15 holds row M=m; half=lane>>4 selects K {8h..8h+7, 16+8h..23+8h}.
// => two contiguous 16B loads.
__device__ __forceinline__ v16bf frag_a(const __bf16* p, int ld, int lane) {
  const int m = lane & 15, h = lane >> 4;
  const __bf16* r = p + m * ld + 8 * h;
  v16bf v;
  reinterpret_cast<uint4*>(&v)[0] = *reinterpret_cast<const uint4*>(r);
  reinterpret_cast<uint4*>(&v)[1] = *reinterpret_cast<const uint4*>(r + 16);
  return v;
}

// B fragment (32x16 bf16, K=rows, N=cols) from TRANSPOSED [n][k] LDS storage.
// Lane n=lane&15 holds column N=n; half selects K {8h..8h+7, 16+8h..23+8h}.
__device__ __forceinline__ v16bf frag_bT(const __bf16* p, int ld, int lane) {
  const int n = lane & 15, h = lane >> 4;
  const __bf16* r = p + n * ld + 8 * h;
  v16bf v;
  reinterpret_cast<uint4*>(&v)[0] = *reinterpret_cast<const uint4*>(r);
  reinterpret_cast<uint4*>(&v)[1] = *reinterpret_cast<const uint4*>(r + 16);
  return v;
}

__device__ __forceinline__ v8f wmma_bf16(v16bf a, v16bf b, v8f c) {
  return __builtin_amdgcn_wmma_f32_16x16x32_bf16(false, a, false, b,
                                                 (short)0, c, false, false);
}

__global__ __launch_bounds__(NTHR)
void maf_fused(const float* __restrict__ x,
               const float* __restrict__ w1,
               const float* __restrict__ w2,
               const float* __restrict__ w3,
               const unsigned char* __restrict__ m1,
               const unsigned char* __restrict__ m2,
               const unsigned char* __restrict__ m3,
               float* __restrict__ out) {
  __shared__ float  s_xr[TB * XFLD];
  __shared__ __bf16 s_xb[TB * XLD];
  __shared__ __align__(16) unsigned char s_wraw[WBUF_BYTES]; // W1T/W2T-chunk/W3T/out
  __shared__ __bf16 s_h[TB * HLD];

  __bf16* wbuf = reinterpret_cast<__bf16*>(s_wraw);

  const int nb   = blockIdx.x;        // network index
  const int b0   = blockIdx.y * TB;   // batch row base
  const int tid  = threadIdx.x;
  const int lane = tid & 31;
  const int wv   = tid >> 5;
  const int R0   = wv * 16;           // this wave's 16-row strip
  const int cn   = lane & 15;         // fragment column/row-in-strip
  const int ch   = lane >> 4;

  // ---- stage x tile (fp32 for epilogue, bf16 as WMMA A) ----
  for (int e = tid; e < TB * D_; e += NTHR) {
    const int r = e >> 5, c = e & 31;
    const float v = x[(size_t)(b0 + r) * F_ + c];
    s_xr[r * XFLD + c] = v;
    s_xb[r * XLD + c]  = (__bf16)v;
  }
  // ---- stage masked W1^T (bf16) ----
  {
    const float* wp = w1 + (size_t)nb * D_ * HID;
    const unsigned char* mp = m1 + (size_t)nb * D_ * HID;
    for (int e = tid; e < D_ * HID; e += NTHR) {
      const int k = e >> 8, n = e & 255;
      const float v = mp[e] ? wp[e] : 0.0f;
      wbuf[n * W1LD + k] = (__bf16)v;
    }
  }
  __syncthreads();

  v8f acc[16];

  // ---- GEMM1: h1 = relu(x[16x32] * W1[32x256]) ----
  {
    const v16bf a = frag_a(s_xb + R0 * XLD, XLD, lane);
#pragma unroll
    for (int nt = 0; nt < 16; ++nt) {
      v8f c = {};
      acc[nt] = wmma_bf16(a, frag_bT(wbuf + nt * 16 * W1LD, W1LD, lane), c);
    }
  }
  // relu -> bf16 h1 (own rows only), zero accumulators for GEMM2
#pragma unroll
  for (int nt = 0; nt < 16; ++nt) {
#pragma unroll
    for (int r = 0; r < 8; ++r) {
      float v = acc[nt][r];
      v = v > 0.0f ? v : 0.0f;
      s_h[(R0 + r + 8 * ch) * HLD + nt * 16 + cn] = (__bf16)v;
      acc[nt][r] = 0.0f;
    }
  }

  // ---- GEMM2: h2 = relu(h1[16x256] * W2[256x256]), W2 staged in 4 K-chunks ----
  {
    const float* w2p = w2 + (size_t)nb * HID * HID;
    const unsigned char* m2p = m2 + (size_t)nb * HID * HID;
    for (int q = 0; q < 4; ++q) {
      __syncthreads();  // previous wbuf contents no longer needed by any wave
      for (int e = tid; e < 64 * HID; e += NTHR) {
        const int kl = e >> 8, n = e & 255;
        const size_t gi = (size_t)(q * 64 + kl) * HID + n;
        const float v = m2p[gi] ? w2p[gi] : 0.0f;
        wbuf[n * W2LD + kl] = (__bf16)v;
      }
      __syncthreads();
#pragma unroll
      for (int kt = 0; kt < 2; ++kt) {
        const v16bf a = frag_a(s_h + R0 * HLD + q * 64 + kt * 32, HLD, lane);
#pragma unroll
        for (int nt = 0; nt < 16; ++nt)
          acc[nt] = wmma_bf16(a, frag_bT(wbuf + nt * 16 * W2LD + kt * 32, W2LD, lane), acc[nt]);
      }
    }
  }
  __syncthreads();  // all waves done reading last W2 chunk

  // relu -> bf16 h2 over h1 (each wave touches only its own 16 rows)
#pragma unroll
  for (int nt = 0; nt < 16; ++nt) {
#pragma unroll
    for (int r = 0; r < 8; ++r) {
      float v = acc[nt][r];
      v = v > 0.0f ? v : 0.0f;
      s_h[(R0 + r + 8 * ch) * HLD + nt * 16 + cn] = (__bf16)v;
    }
  }

  // ---- stage masked W3^T ----
  {
    const float* wp = w3 + (size_t)nb * HID * (2 * D_);
    const unsigned char* mp = m3 + (size_t)nb * HID * (2 * D_);
    for (int e = tid; e < HID * 64; e += NTHR) {
      const int k = e >> 6, n = e & 63;
      const float v = mp[e] ? wp[e] : 0.0f;
      wbuf[n * W3LD + k] = (__bf16)v;
    }
  }
  __syncthreads();

  // ---- GEMM3: out = h2[16x256] * W3[256x64] ----
  v8f acc3[4];
#pragma unroll
  for (int nt = 0; nt < 4; ++nt) acc3[nt] = v8f{};
#pragma unroll
  for (int kt = 0; kt < 8; ++kt) {
    const v16bf a = frag_a(s_h + R0 * HLD + kt * 32, HLD, lane);
#pragma unroll
    for (int nt = 0; nt < 4; ++nt)
      acc3[nt] = wmma_bf16(a, frag_bT(wbuf + nt * 16 * W3LD + kt * 32, W3LD, lane), acc3[nt]);
  }
  __syncthreads();  // all waves done reading W3 before out-tile overlays wbuf

  // ---- spill out tile (fp32) to LDS; shift/log_scale are adjacent columns ----
  float* outt = reinterpret_cast<float*>(s_wraw);
#pragma unroll
  for (int nt = 0; nt < 4; ++nt)
#pragma unroll
    for (int r = 0; r < 8; ++r)
      outt[(R0 + r + 8 * ch) * OLD + nt * 16 + cn] = acc3[nt][r];
  // Each wave reads only rows it just wrote: no barrier needed.

  // ---- Gaussian log-likelihood reduction over D=32 dims ----
  {
    const int lrow = cn;                 // local row in strip
    const int dh   = ch;                 // which half of the 32 dims
    const float* ob = outt + (R0 + lrow) * OLD + dh * 32;
    const float* xb = s_xr + (R0 + lrow) * XFLD + dh * 16;
    float s = 0.0f;
#pragma unroll
    for (int i = 0; i < 16; ++i) {
      const float shift = ob[2 * i];
      const float logs  = ob[2 * i + 1];
      const float yv    = (xb[i] - shift) * __expf(-logs);
      s += -0.5f * yv * yv - HALF_LOG_2PI - logs;
    }
    s += __shfl_xor(s, 16, 32);          // combine the two d-halves
    if (dh == 0)
      out[(size_t)(b0 + R0 + lrow) * NNET + nb] = s;
  }
}

} // anonymous namespace

extern "C" void kernel_launch(void* const* d_in, const int* in_sizes, int n_in,
                              void* d_out, int out_size, void* d_ws, size_t ws_size,
                              hipStream_t stream) {
  (void)in_sizes; (void)n_in; (void)out_size; (void)d_ws; (void)ws_size;
  const float* x  = (const float*)d_in[0];
  const float* w1 = (const float*)d_in[1];
  const float* w2 = (const float*)d_in[2];
  const float* w3 = (const float*)d_in[3];
  const unsigned char* m1 = (const unsigned char*)d_in[4];
  const unsigned char* m2 = (const unsigned char*)d_in[5];
  const unsigned char* m3 = (const unsigned char*)d_in[6];

  dim3 grid(NNET, BTOT / TB);           // 64 networks x 32 row-tiles
  maf_fused<<<grid, NTHR, 0, stream>>>(x, w1, w2, w3, m1, m2, m3, (float*)d_out);
}
